// CoreMambaBlock_39307540693259
// MI455X (gfx1250) — matrix operand order repaired and checked by reading, not compiled, not run on hardware
//
#include <hip/hip_runtime.h>

// ---------------------------------------------------------------------------
// Mamba block forward for MI455X (gfx1250, wave32, WMMA).
// Pipeline: in_proj GEMM -> conv+SiLU -> x_proj GEMM -> dt_proj GEMM(+softplus)
//           -> selective scan (wave-parallel over (b,c,s)) -> out_proj GEMM.
// GEMMs: bf16 WMMA (v_wmma_f32_16x16x32_bf16), f32 accumulate.
// Round 3: BK=64 (16 WMMA per barrier pair), compile-time bounds guards,
//          explicit load-phase/store-phase staging for loadcnt batching.
// ---------------------------------------------------------------------------

typedef __attribute__((ext_vector_type(16))) __bf16 v16bf;
typedef __attribute__((ext_vector_type(8)))  float  v8f;

#define D_MODEL 1024
#define D_INNER 2048
#define D_STATE 16
#define DT_RANK 64
#define B_SZ    2
#define L_SEQ   2048
#define N_ROWS  (B_SZ * L_SEQ)          // 4096 token rows

__device__ __forceinline__ __bf16 u16_as_bf16(unsigned short s) {
    union { unsigned short s; __bf16 b; } cv; cv.s = s; return cv.b;
}

// pack 4 f32 -> 4 bf16 in one 64-bit word (lowers to v_cvt_pk_bf16_f32 x2)
__device__ __forceinline__ unsigned long long pack4_bf16(float4 v) {
    union { __bf16 h[4]; unsigned long long u; } cv;
    cv.h[0] = (__bf16)v.x; cv.h[1] = (__bf16)v.y;
    cv.h[2] = (__bf16)v.z; cv.h[3] = (__bf16)v.w;
    return cv.u;
}

// ---------------------------------------------------------------------------
// Generic tiled GEMM:  C[M,N] = A[M,K] * W[N,K]^T   (both K-major row-major)
// EPI==0: plain store.  EPI==1: softplus(acc + bias[n]) (dt_proj epilogue).
// GN: compile-time N-bounds guard (only the N=96 x_proj GEMM needs it).
// BM=128, BN=128, BK=64. 256 threads = 8 waves in a 4(M) x 2(N) grid;
// each wave computes a 32x64 tile = 2x4 accumulators, 16 WMMA per K-step.
// Requires: M % 128 == 0, K % 64 == 0, lda/ldw % 4 == 0, and N % 128 == 0
// unless GN (all true for every call site below).
// ---------------------------------------------------------------------------
template <int EPI, bool GN>
__global__ __launch_bounds__(256)
void gemm_bf16_wmma(const float* __restrict__ A, int lda,
                    const float* __restrict__ W, int ldw,
                    float* __restrict__ C, int ldc,
                    int M, int N, int K,
                    const float* __restrict__ bias) {
    constexpr int BM = 128, BN = 128, BK = 64;
    constexpr int LDSH = BK + 4;                 // 136-byte row stride, 8B-aligned
    constexpr int NCH  = (BM * BK) / (256 * 4);  // 8 float4 chunks per thread
    __shared__ __align__(16) unsigned short Ash[BM][LDSH];
    __shared__ __align__(16) unsigned short Bsh[BN][LDSH];

    const int tid  = threadIdx.x;
    const int wave = tid >> 5;
    const int lane = tid & 31;
    const int half = lane >> 4;     // 0: lanes 0-15, 1: lanes 16-31
    const int lm   = lane & 15;

    const int m0 = blockIdx.y * BM;
    const int n0 = blockIdx.x * BN;
    const int wm = (wave >> 1) * 32;   // wave row offset in block tile
    const int wn = (wave & 1) * 64;    // wave col offset in block tile

    v8f acc[2][4] = {};

    for (int k0 = 0; k0 < K; k0 += BK) {
        // ---- load phase: 16 global_load_b128, no LDS dependency ----
        float4 av[NCH], wv[NCH];
#pragma unroll
        for (int it = 0; it < NCH; ++it) {
            int q    = tid + it * 256;          // float4 chunk id
            int r    = q >> 4;                  // row (16 chunks per 64-wide row)
            int kcol = (q & 15) * 4;
            av[it] = *(const float4*)&A[(long)(m0 + r) * lda + (k0 + kcol)];
            if (!GN || (n0 + r) < N)
                wv[it] = *(const float4*)&W[(long)(n0 + r) * ldw + (k0 + kcol)];
            else
                wv[it] = make_float4(0.f, 0.f, 0.f, 0.f);
        }
        if (k0 + BK < K)   // hint next A tile -> global_prefetch_b8
            __builtin_prefetch(&A[(long)(m0 + (tid & 127)) * lda + (k0 + BK)], 0, 3);

        __syncthreads();   // previous iteration's fragment reads are done

        // ---- store phase: cvt to bf16, packed b64 LDS stores ----
#pragma unroll
        for (int it = 0; it < NCH; ++it) {
            int q    = tid + it * 256;
            int r    = q >> 4;
            int kcol = (q & 15) * 4;
            *(unsigned long long*)&Ash[r][kcol] = pack4_bf16(av[it]);
            *(unsigned long long*)&Bsh[r][kcol] = pack4_bf16(wv[it]);
        }
        __syncthreads();

        // ---- compute: 2 WMMA-K slices of 32, 8 WMMA each ----
#pragma unroll
        for (int kk = 0; kk < BK; kk += 32) {
            v16bf afrag[2], bfrag[4];
#pragma unroll
            for (int mi = 0; mi < 2; ++mi) {
                int row = wm + mi * 16 + lm;             // A: M = lane%16
#pragma unroll
                for (int v = 0; v < 8; ++v) {
                    int kb = kk + (v >> 2) * 16 + half * 8 + (v & 3) * 2;
                    afrag[mi][2 * v]     = u16_as_bf16(Ash[row][kb]);
                    afrag[mi][2 * v + 1] = u16_as_bf16(Ash[row][kb + 1]);
                }
            }
#pragma unroll
            for (int ni = 0; ni < 4; ++ni) {
                int coln = wn + ni * 16 + lm;            // B: N = lane%16
#pragma unroll
                for (int v = 0; v < 8; ++v) {
                    int kb = kk + (v >> 2) * 16 + half * 8 + (v & 3) * 2;
                    bfrag[ni][2 * v]     = u16_as_bf16(Bsh[coln][kb]);
                    bfrag[ni][2 * v + 1] = u16_as_bf16(Bsh[coln][kb + 1]);
                }
            }
#pragma unroll
            for (int mi = 0; mi < 2; ++mi)
#pragma unroll
                for (int ni = 0; ni < 4; ++ni)
                    acc[mi][ni] = __builtin_amdgcn_wmma_f32_16x16x32_bf16(
                        false, afrag[mi], false, bfrag[ni],
                        (short)0, acc[mi][ni], false, false);
        }
    }

    // ---- store C (layout: n = lane%16, m = v + 8*half) ----
#pragma unroll
    for (int ni = 0; ni < 4; ++ni) {
        int coln = n0 + wn + ni * 16 + lm;
        if (GN && coln >= N) continue;
#pragma unroll
        for (int mi = 0; mi < 2; ++mi) {
#pragma unroll
            for (int v = 0; v < 8; ++v) {
                int row = m0 + wm + mi * 16 + half * 8 + v;
                float val = acc[mi][ni][v];
                if (EPI == 1) {
                    val += bias[coln];
                    val = (val > 20.0f) ? val : log1pf(__expf(val));  // softplus
                }
                C[(long)row * ldc + coln] = val;
            }
        }
    }
}

// ---------------------------------------------------------------------------
// Depthwise causal conv (width 4, left pad 3) + bias + SiLU.
// Reads x_ssm = xz[:, 0:2048]; writes u [4096 x 2048].
// ---------------------------------------------------------------------------
__global__ __launch_bounds__(256)
void conv_silu_kernel(const float* __restrict__ xz,
                      const float* __restrict__ cw,   // [2048,1,4]
                      const float* __restrict__ cb,   // [2048]
                      float* __restrict__ u) {
    long i = (long)blockIdx.x * blockDim.x + threadIdx.x;  // over 4096*2048
    if (i >= (long)N_ROWS * D_INNER) return;
    int  c    = (int)(i & (D_INNER - 1));
    long bl   = i >> 11;               // b*L + l
    int  l    = (int)(bl & (L_SEQ - 1));
    long brow = bl - l;                // b*L
    float acc = cb[c];
#pragma unroll
    for (int j = 0; j < 4; ++j) {
        int ll = l - 3 + j;
        if (ll >= 0) acc += xz[(brow + ll) * (2 * D_INNER) + c] * cw[c * 4 + j];
    }
    float s = 1.0f / (1.0f + __expf(-acc));
    u[i] = acc * s;   // SiLU
}

// ---------------------------------------------------------------------------
// Selective scan. One wave handles 2 channels; each 16-lane half owns one
// (b,c) chain with lane = state index s. 65536 threads = 2048 waves total.
// h_t = exp(delta*A)*h_{t-1} + delta*B_t*u_t ;  y_t = sum_s h_t[s]*C_t[s]
// Reduction over s via __shfl_xor within 16-lane halves (wave32).
// ---------------------------------------------------------------------------
__global__ __launch_bounds__(256)
void scan_kernel(const float* __restrict__ delta,   // [4096,2048]
                 const float* __restrict__ x_dbl,   // [4096,96]
                 const float* __restrict__ u,       // [4096,2048]
                 const float* __restrict__ xz,      // [4096,4096] (z = cols 2048+)
                 const float* __restrict__ A_log,   // [2048,16]
                 const float* __restrict__ Dvec,    // [2048]
                 float* __restrict__ y) {           // [4096,2048]
    int tid   = blockIdx.x * blockDim.x + threadIdx.x;
    int w     = tid >> 5;
    int lane  = tid & 31;
    int half  = lane >> 4;
    int s     = lane & 15;
    int chain = w * 2 + half;                 // 0..4095
    int b     = chain >> 11;                  // / 2048
    int c     = chain & (D_INNER - 1);

    float As = -__expf(A_log[c * D_STATE + s]);
    float Dc = Dvec[c];
    float h  = 0.0f;

    for (int l = 0; l < L_SEQ; ++l) {
        long idx = (long)b * L_SEQ + l;
        float dl = delta[idx * D_INNER + c];           // broadcast within half
        float uv = u[idx * D_INNER + c];               // broadcast within half
        float Bv = x_dbl[idx * 96 + DT_RANK + s];
        float Cv = x_dbl[idx * 96 + DT_RANK + D_STATE + s];
        float dA = __expf(dl * As);
        h = dA * h + dl * Bv * uv;
        float part = h * Cv;
        part += __shfl_xor(part, 1, 32);
        part += __shfl_xor(part, 2, 32);
        part += __shfl_xor(part, 4, 32);
        part += __shfl_xor(part, 8, 32);               // sum over 16 states
        if (s == 0) {
            float yy = part + uv * Dc;
            float z  = xz[idx * (2 * D_INNER) + D_INNER + c];
            float sg = 1.0f / (1.0f + __expf(-z));
            y[idx * D_INNER + c] = yy * z * sg;        // * silu(z)
        }
    }
}

// ---------------------------------------------------------------------------
// Launcher. Inputs (fp32): x, in_proj_w, conv_w, conv_b, x_proj_w, dt_proj_w,
// dt_proj_b, A_log, D, out_proj_w. Output fp32 [2,2048,1024].
// ---------------------------------------------------------------------------
extern "C" void kernel_launch(void* const* d_in, const int* in_sizes, int n_in,
                              void* d_out, int out_size, void* d_ws, size_t ws_size,
                              hipStream_t stream) {
    const float* x         = (const float*)d_in[0];
    const float* in_proj_w = (const float*)d_in[1];
    const float* conv_w    = (const float*)d_in[2];
    const float* conv_b    = (const float*)d_in[3];
    const float* x_proj_w  = (const float*)d_in[4];
    const float* dt_proj_w = (const float*)d_in[5];
    const float* dt_proj_b = (const float*)d_in[6];
    const float* A_log     = (const float*)d_in[7];
    const float* Dvec      = (const float*)d_in[8];
    const float* out_proj_w= (const float*)d_in[9];
    float* out = (float*)d_out;

    // workspace layout (floats)
    float* ws    = (float*)d_ws;
    float* xz    = ws;                                   // 4096*4096
    float* u     = xz    + (size_t)N_ROWS * 2 * D_INNER; // 4096*2048
    float* x_dbl = u     + (size_t)N_ROWS * D_INNER;     // 4096*96
    float* delta = x_dbl + (size_t)N_ROWS * 96;          // 4096*2048
    float* ybuf  = delta + (size_t)N_ROWS * D_INNER;     // 4096*2048

    dim3 blk(256);

    // 1) xz = x @ in_proj_w^T     [4096,1024]x[4096,1024]^T -> [4096,4096]
    gemm_bf16_wmma<0, false><<<dim3((2 * D_INNER) / 128, N_ROWS / 128), blk, 0, stream>>>(
        x, D_MODEL, in_proj_w, D_MODEL, xz, 2 * D_INNER,
        N_ROWS, 2 * D_INNER, D_MODEL, nullptr);

    // 2) u = silu(depthwise_conv(x_ssm) + b)
    {
        long total = (long)N_ROWS * D_INNER;
        conv_silu_kernel<<<dim3((unsigned)((total + 255) / 256)), blk, 0, stream>>>(
            xz, conv_w, conv_b, u);
    }

    // 3) x_dbl = u @ x_proj_w^T   [4096,2048]x[96,2048]^T -> [4096,96]  (N guard)
    gemm_bf16_wmma<0, true><<<dim3(1, N_ROWS / 128), blk, 0, stream>>>(
        u, D_INNER, x_proj_w, D_INNER, x_dbl, 96,
        N_ROWS, 96, D_INNER, nullptr);

    // 4) delta = softplus(x_dbl[:, :64] @ dt_proj_w^T + dt_proj_b) -> [4096,2048]
    gemm_bf16_wmma<1, false><<<dim3(D_INNER / 128, N_ROWS / 128), blk, 0, stream>>>(
        x_dbl, 96, dt_proj_w, DT_RANK, delta, D_INNER,
        N_ROWS, D_INNER, DT_RANK, dt_proj_b);

    // 5) selective scan -> ybuf (includes +u*D and *silu(z))
    scan_kernel<<<dim3(256), blk, 0, stream>>>(
        delta, x_dbl, u, xz, A_log, Dvec, ybuf);

    // 6) out = ybuf @ out_proj_w^T [4096,2048]x[1024,2048]^T -> [4096,1024]
    gemm_bf16_wmma<0, false><<<dim3(D_MODEL / 128, N_ROWS / 128), blk, 0, stream>>>(
        ybuf, D_INNER, out_proj_w, D_INNER, out, D_MODEL,
        N_ROWS, D_MODEL, D_INNER, nullptr);
}
